// GATLSTM_44676249813673
// MI455X (gfx1250) — compile-verified, matrix operand order checked by reading, not run on hardware
//
#include <hip/hip_runtime.h>
#include <hip/hip_bf16.h>
#include <math.h>

namespace {

constexpr int Bv = 8, Tv = 128, Nv = 512, HIDv = 4;
constexpr int Dv   = Nv * HIDv;      // 2048
constexpr int Ev   = 16384;
constexpr int BNv  = Bv * Nv;        // 4096
constexpr int G4D  = 4 * Dv;         // 8192
constexpr int ROWS = Tv * Bv;        // 1024
constexpr int NEDGE = Ev + BNv;      // 20480 (edges + self loops)
constexpr float NEG_SLOPE = 0.2f;

typedef __attribute__((ext_vector_type(16))) __bf16 v16bf;
typedef __attribute__((ext_vector_type(8)))  float  v8f;

union Acc8 { v8f v; float f[8]; };

// order-preserving float <-> uint encoding for atomicMax on floats
__device__ __forceinline__ unsigned f2ord(float f) {
  unsigned u = __float_as_uint(f);
  return (u & 0x80000000u) ? ~u : (u | 0x80000000u);
}
__device__ __forceinline__ float ord2f(unsigned u) {
  return (u & 0x80000000u) ? __uint_as_float(u & 0x7FFFFFFFu)
                           : __uint_as_float(~u);
}

// 16x32 bf16 fragment: lane l<16 -> row l, K=[k0,k0+16); lane>=16 -> row l-16, K=[k0+16,k0+32)
__device__ __forceinline__ v16bf load_frag_bf16(const __hip_bfloat16* base,
                                                int ld, int r0, int k0) {
  const int lane = threadIdx.x & 31;
  const int row  = lane & 15;
  const int kh   = (lane >> 4) << 4;
  return *reinterpret_cast<const v16bf*>(base + (size_t)(r0 + row) * ld + (k0 + kh));
}

// ---------------- init ----------------
__global__ void k_init(unsigned* m_u, float* denom, float* aggn,
                       __hip_bfloat16* hb, float* c) {
  int i = blockIdx.x * blockDim.x + threadIdx.x;
  if (i < Tv * BNv) {
    m_u[i]   = f2ord(-__builtin_inff());
    denom[i] = 0.f;
    aggn[i]  = 0.f;
  }
  if (i < 16 * Dv) hb[i] = __float2bfloat16(0.f);   // padded h (rows 8..15 stay 0)
  if (i < Bv * Dv) c[i] = 0.f;
}

__global__ void k_cvt_bf16(const float* src, __hip_bfloat16* dst, int n) {
  int i = blockIdx.x * blockDim.x + threadIdx.x;
  if (i < n) dst[i] = __float2bfloat16(src[i]);
}

// ---------------- GAT (collapsed to scalar softmax aggregation) ----------------
__device__ __forceinline__ void edge_score(const float* x, const int* ei,
    const float* gw, const float* as_, const float* ad_,
    int t, int e, int& dst, float& escore, float& xsrc) {
  int s, d;
  if (e < Ev) { s = ei[e]; d = ei[Ev + e]; } else { s = d = e - Ev; }
  float Ss = 0.f, Sd = 0.f;
#pragma unroll
  for (int k = 0; k < HIDv; ++k) { Ss += gw[k] * as_[k]; Sd += gw[k] * ad_[k]; }
  float xs = x[(s / Nv) * (Tv * Nv) + t * Nv + (s % Nv)];
  float xd = x[(d / Nv) * (Tv * Nv) + t * Nv + (d % Nv)];
  float v = xs * Ss + xd * Sd;
  escore = (v > 0.f) ? v : NEG_SLOPE * v;
  dst = d; xsrc = xs;
}

__global__ void k_gat_max(const float* x, const int* ei, const float* gw,
                          const float* as_, const float* ad_, unsigned* m_u) {
  int gid = blockIdx.x * blockDim.x + threadIdx.x;
  if (gid >= Tv * NEDGE) return;
  int t = gid / NEDGE, e = gid % NEDGE;
  int dst; float es, xs;
  edge_score(x, ei, gw, as_, ad_, t, e, dst, es, xs);
  atomicMax(&m_u[t * BNv + dst], f2ord(es));
}

__global__ void k_gat_sum(const float* x, const int* ei, const float* gw,
                          const float* as_, const float* ad_,
                          const unsigned* m_u, float* denom, float* aggn) {
  int gid = blockIdx.x * blockDim.x + threadIdx.x;
  if (gid >= Tv * NEDGE) return;
  int t = gid / NEDGE, e = gid % NEDGE;
  int dst; float es, xs;
  edge_score(x, ei, gw, as_, ad_, t, e, dst, es, xs);
  float ex = __expf(es - ord2f(m_u[t * BNv + dst]));
  atomicAdd(&denom[t * BNv + dst], ex);
  atomicAdd(&aggn[t * BNv + dst], ex * xs);
}

__global__ void k_gat_fin(const float* denom, const float* aggn,
                          const float* gw, const float* gbias,
                          __hip_bfloat16* seqb) {
  int gid = blockIdx.x * blockDim.x + threadIdx.x;
  if (gid >= Tv * BNv) return;
  int t = gid / BNv, i = gid % BNv;
  int b = i / Nv, n = i % Nv;
  float a = aggn[gid] / denom[gid];   // self-loops guarantee denom > 0
#pragma unroll
  for (int k = 0; k < HIDv; ++k) {
    float v = gw[k] * a + gbias[k];
    v = v > 0.f ? v : 0.f;            // relu
    seqb[(size_t)t * (Bv * Dv) + (size_t)b * Dv + n * HIDv + k] = __float2bfloat16(v);
  }
}

// ---------------- GEMMs (WMMA bf16 -> f32) ----------------
// Zx[1024,8192] = seq[1024,2048] @ w_ih[8192,2048]^T + (b_ih + b_hh)
// 2x2 register blocking: each wave owns a 32x32 output block (4 WMMA tiles),
// so 4 fragment loads feed 4 WMMAs per K-step (2x arithmetic intensity).
__global__ void k_gemm_zx(const __hip_bfloat16* seqb, const __hip_bfloat16* wihb,
                          const float* b_ih, const float* b_hh, float* Zx) {
  const int NT2 = G4D / 32;   // 256 n-blocks
  const int MT2 = ROWS / 32;  // 32 m-blocks
  int tile = blockIdx.x * (blockDim.x >> 5) + (threadIdx.x >> 5);
  if (tile >= MT2 * NT2) return;
  int m0 = (tile / NT2) * 32, n0 = (tile % NT2) * 32;
  Acc8 acc[2][2];
#pragma unroll
  for (int mi = 0; mi < 2; ++mi)
#pragma unroll
    for (int ni = 0; ni < 2; ++ni)
#pragma unroll
      for (int r = 0; r < 8; ++r) acc[mi][ni].f[r] = 0.f;

  for (int k0 = 0; k0 < Dv; k0 += 32) {
    v16bf a0 = load_frag_bf16(seqb, Dv, m0,      k0);
    v16bf a1 = load_frag_bf16(seqb, Dv, m0 + 16, k0);
    v16bf b0 = load_frag_bf16(wihb, Dv, n0,      k0);
    v16bf b1 = load_frag_bf16(wihb, Dv, n0 + 16, k0);
    __builtin_prefetch(wihb + (size_t)n0 * Dv + k0 + 256, 0, 1);
    acc[0][0].v = __builtin_amdgcn_wmma_f32_16x16x32_bf16(
        false, a0, false, b0, (short)0, acc[0][0].v, false, false);
    acc[0][1].v = __builtin_amdgcn_wmma_f32_16x16x32_bf16(
        false, a0, false, b1, (short)0, acc[0][1].v, false, false);
    acc[1][0].v = __builtin_amdgcn_wmma_f32_16x16x32_bf16(
        false, a1, false, b0, (short)0, acc[1][0].v, false, false);
    acc[1][1].v = __builtin_amdgcn_wmma_f32_16x16x32_bf16(
        false, a1, false, b1, (short)0, acc[1][1].v, false, false);
  }
  int lane = threadIdx.x & 31;
  int col = lane & 15, rbase = (lane >> 4) << 3;
#pragma unroll
  for (int mi = 0; mi < 2; ++mi) {
#pragma unroll
    for (int ni = 0; ni < 2; ++ni) {
#pragma unroll
      for (int r = 0; r < 8; ++r) {
        int row = m0 + mi * 16 + rbase + r;
        int j   = n0 + ni * 16 + col;
        Zx[(size_t)row * G4D + j] = acc[mi][ni].f[r] + b_ih[j] + b_hh[j];
      }
    }
  }
}

// z[8,8192] = Zx[t] + h[16(pad),2048] @ w_hh[8192,2048]^T
// Kept unblocked: each serialized step is latency-bound; 512 independent waves
// maximize WGP coverage, and w_hh traffic (32MB/step from L2) is tile-exact.
__global__ void k_gemm_hh(const __hip_bfloat16* hb, const __hip_bfloat16* whhb,
                          const float* Zx, int t, float* z) {
  const int NT = G4D / 16;  // 512
  int tile = blockIdx.x * (blockDim.x >> 5) + (threadIdx.x >> 5);
  if (tile >= NT) return;
  int n0 = tile * 16;
  Acc8 acc;
#pragma unroll
  for (int r = 0; r < 8; ++r) acc.f[r] = 0.f;
  for (int k0 = 0; k0 < Dv; k0 += 32) {
    v16bf a  = load_frag_bf16(hb, Dv, 0, k0);
    v16bf bb = load_frag_bf16(whhb, Dv, n0, k0);
    __builtin_prefetch(whhb + (size_t)n0 * Dv + k0 + 256, 0, 1);
    acc.v = __builtin_amdgcn_wmma_f32_16x16x32_bf16(
        false, a, false, bb, (short)0, acc.v, false, false);
  }
  int lane = threadIdx.x & 31;
  int col = lane & 15, rbase = (lane >> 4) << 3;
#pragma unroll
  for (int r = 0; r < 8; ++r) {
    int row = rbase + r;
    if (row < Bv) {
      int j = n0 + col;
      z[(size_t)row * G4D + j] = acc.f[r] + Zx[(size_t)(t * Bv + row) * G4D + j];
    }
  }
}

// LSTM gate update: c' = sig(f)*c + sig(i)*tanh(g); h' = sig(o)*tanh(c')
__global__ void k_gate(const float* z, float* c, __hip_bfloat16* hb) {
  int idx = blockIdx.x * blockDim.x + threadIdx.x;
  if (idx >= Bv * Dv) return;
  int b = idx / Dv, d = idx % Dv;
  const float* zb = z + (size_t)b * G4D;
  float zi = zb[d], zf = zb[Dv + d], zg = zb[2 * Dv + d], zo = zb[3 * Dv + d];
  float si = 1.f / (1.f + __expf(-zi));
  float sf = 1.f / (1.f + __expf(-zf));
  float so = 1.f / (1.f + __expf(-zo));
  float cn = sf * c[idx] + si * tanhf(zg);
  c[idx] = cn;
  hb[(size_t)b * Dv + d] = __float2bfloat16(so * tanhf(cn));
}

// out[8,512] = hT[16(pad),2048] @ w_lin[512,2048]^T + b_lin
__global__ void k_gemm_out(const __hip_bfloat16* hb, const __hip_bfloat16* wlinb,
                           const float* b_lin, float* out) {
  const int NT = Nv / 16;  // 32
  int tile = blockIdx.x * (blockDim.x >> 5) + (threadIdx.x >> 5);
  if (tile >= NT) return;
  int n0 = tile * 16;
  Acc8 acc;
#pragma unroll
  for (int r = 0; r < 8; ++r) acc.f[r] = 0.f;
  for (int k0 = 0; k0 < Dv; k0 += 32) {
    v16bf a  = load_frag_bf16(hb, Dv, 0, k0);
    v16bf bb = load_frag_bf16(wlinb, Dv, n0, k0);
    acc.v = __builtin_amdgcn_wmma_f32_16x16x32_bf16(
        false, a, false, bb, (short)0, acc.v, false, false);
  }
  int lane = threadIdx.x & 31;
  int col = lane & 15, rbase = (lane >> 4) << 3;
#pragma unroll
  for (int r = 0; r < 8; ++r) {
    int row = rbase + r;
    if (row < Bv) out[(size_t)row * Nv + n0 + col] = acc.f[r] + b_lin[n0 + col];
  }
}

} // namespace

extern "C" void kernel_launch(void* const* d_in, const int* in_sizes, int n_in,
                              void* d_out, int out_size, void* d_ws, size_t ws_size,
                              hipStream_t stream) {
  (void)in_sizes; (void)n_in; (void)out_size; (void)ws_size;
  const float* x     = (const float*)d_in[0];
  const float* gw    = (const float*)d_in[1];
  const float* asrc  = (const float*)d_in[2];
  const float* adst  = (const float*)d_in[3];
  const float* gbias = (const float*)d_in[4];
  const float* wih   = (const float*)d_in[5];
  const float* whh   = (const float*)d_in[6];
  const float* bih   = (const float*)d_in[7];
  const float* bhh   = (const float*)d_in[8];
  const float* wlin  = (const float*)d_in[9];
  const float* blin  = (const float*)d_in[10];
  const int*   ei    = (const int*)d_in[11];
  float* out = (float*)d_out;

  char* ws = (char*)d_ws;
  size_t off = 0;
  auto take = [&](size_t bytes) -> char* {
    char* p = ws + off;
    off += (bytes + 255) & ~(size_t)255;
    return p;
  };
  unsigned*        m_u   = (unsigned*)       take((size_t)Tv * BNv * 4);
  float*           denom = (float*)          take((size_t)Tv * BNv * 4);
  float*           aggn  = (float*)          take((size_t)Tv * BNv * 4);
  __hip_bfloat16*  seqb  = (__hip_bfloat16*) take((size_t)ROWS * Dv * 2);
  __hip_bfloat16*  wihb  = (__hip_bfloat16*) take((size_t)G4D * Dv * 2);
  __hip_bfloat16*  whhb  = (__hip_bfloat16*) take((size_t)G4D * Dv * 2);
  __hip_bfloat16*  wlinb = (__hip_bfloat16*) take((size_t)Nv * Dv * 2);
  float*           Zx    = (float*)          take((size_t)ROWS * G4D * 4);
  float*           zbuf  = (float*)          take((size_t)Bv * G4D * 4);
  float*           cbuf  = (float*)          take((size_t)Bv * Dv * 4);
  __hip_bfloat16*  hb    = (__hip_bfloat16*) take((size_t)16 * Dv * 2);

  const int TPB = 256;

  k_init<<<(Tv * BNv + TPB - 1) / TPB, TPB, 0, stream>>>(m_u, denom, aggn, hb, cbuf);

  k_cvt_bf16<<<(G4D * Dv + TPB - 1) / TPB, TPB, 0, stream>>>(wih, wihb, G4D * Dv);
  k_cvt_bf16<<<(G4D * Dv + TPB - 1) / TPB, TPB, 0, stream>>>(whh, whhb, G4D * Dv);
  k_cvt_bf16<<<(Nv * Dv + TPB - 1) / TPB, TPB, 0, stream>>>(wlin, wlinb, Nv * Dv);

  const int edthreads = Tv * NEDGE;
  k_gat_max<<<(edthreads + TPB - 1) / TPB, TPB, 0, stream>>>(x, ei, gw, asrc, adst, m_u);
  k_gat_sum<<<(edthreads + TPB - 1) / TPB, TPB, 0, stream>>>(x, ei, gw, asrc, adst,
                                                             m_u, denom, aggn);
  k_gat_fin<<<(Tv * BNv + TPB - 1) / TPB, TPB, 0, stream>>>(denom, aggn, gw, gbias, seqb);

  // input-projection GEMM: 32 x 256 blocks of 32x32, 4 waves/block
  k_gemm_zx<<<(32 * 256) / 4, 128, 0, stream>>>(seqb, wihb, bih, bhh, Zx);

  // sequential recurrence: 512 tiles/step, 4 waves/block
  for (int t = 0; t < Tv; ++t) {
    k_gemm_hh<<<512 / 4, 128, 0, stream>>>(hb, whhb, Zx, t, zbuf);
    k_gate<<<(Bv * Dv + TPB - 1) / TPB, TPB, 0, stream>>>(zbuf, cbuf, hb);
  }

  // final linear: 32 tiles
  k_gemm_out<<<32 / 4, 128, 0, stream>>>(hb, wlinb, blin, out);
}
